// MultiHeadAttention_62431644614926
// MI455X (gfx1250) — matrix operand order repaired and checked by reading, not compiled, not run on hardware
//
#include <hip/hip_runtime.h>

// ---------- types ----------
typedef __attribute__((ext_vector_type(16))) __bf16 v16bf;
typedef __attribute__((ext_vector_type(8)))  float  v8f;
typedef unsigned int v4u  __attribute__((ext_vector_type(4)));
typedef v4u          v4ua __attribute__((may_alias));
typedef float        v4f  __attribute__((ext_vector_type(4)));
typedef v4f          v4fa __attribute__((may_alias));

union Frag {
  v16bf          bf;
  unsigned short u[16];
  v4u            q[2];
};

// f32 -> bf16 (round to nearest even)
__device__ __forceinline__ unsigned short f2bf(float f) {
  unsigned int u = __float_as_uint(f);
  u += 0x7FFFu + ((u >> 16) & 1u);
  return (unsigned short)(u >> 16);
}

// A-fragment (16x32 bf16): lane holds row lm; K = 16*half .. +15 contiguous
__device__ __forceinline__ void load_afrag(Frag& f, const unsigned short* row, int half) {
  f.q[0] = *(const v4ua*)(row + 16 * half);
  f.q[1] = *(const v4ua*)(row + 16 * half + 8);
}
// B-fragment (32x16 bf16): lane holds col lm; K = 16*half .. +15 contiguous
__device__ __forceinline__ void load_bfrag(Frag& f, const unsigned short* colK) {
  f.q[0] = *(const v4ua*)(colK);
  f.q[1] = *(const v4ua*)(colK + 8);
}

// ==========================================================================
// GEMM: C[M,N] = A[M,K] * W[K,N] + bias[N]
// Block = 256 threads (8 waves); block tile 128(M) x 64(N).
// Wave w owns rows [tileM + 16w, +16), all 64 cols: 4 accumulators, 4 WMMAs
// per 32-deep K-step, one shared LDS B stage (32x64 bf16, padded stride 40).
// ==========================================================================
#define BSTR 40  // LDS column stride (shorts): 80B rows -> 16B aligned, 2-way banks

template <bool A_BF16, bool OUT_F32>
__global__ __launch_bounds__(256) void gemm_wmma(
    const void* __restrict__ Ain, const float* __restrict__ W,
    const float* __restrict__ bias, void* __restrict__ Cout,
    int M, int K, int N)
{
  __shared__ __align__(16) unsigned short ldsB[64 * BSTR]; // [col][k] col-major

  const int tid   = threadIdx.x;
  const int wave  = tid >> 5;
  const int lane  = tid & 31;
  const int lm    = lane & 15;
  const int half  = lane >> 4;
  const int tileN = blockIdx.x * 64;
  const int tileM = blockIdx.y * 128 + wave * 16;
  const int rowA  = tileM + lm;

  v8f acc[4] = {v8f{}, v8f{}, v8f{}, v8f{}};

  for (int kb = 0; kb < K; kb += 32) {
    __syncthreads();
    // stage W[kb..kb+32, tileN..tileN+64] transposed -> ldsB[c*BSTR + k]
#pragma unroll
    for (int i = 0; i < 8; ++i) {
      int e = tid + 256 * i;   // 0..2047
      int k = e >> 6;          // 0..31
      int c = e & 63;          // 0..63
      ldsB[c * BSTR + k] = f2bf(W[(size_t)(kb + k) * N + tileN + c]);
    }
    __syncthreads();

    // A fragment for this wave's 16 rows (shared across the 4 N-tiles)
    Frag fa;
    if (A_BF16) {
      const unsigned short* pa = (const unsigned short*)Ain + (size_t)rowA * K + kb;
      load_afrag(fa, pa, half);
    } else {
      const float* pa = (const float*)Ain + (size_t)rowA * K + kb;
      const v4fa* p0 = (const v4fa*)(pa + 16 * half);
      const v4fa* p1 = (const v4fa*)(pa + 16 * half + 8);
      v4f a0 = p0[0], a1 = p0[1], a2 = p1[0], a3 = p1[1];
#pragma unroll
      for (int i = 0; i < 4; ++i) {
        fa.u[i]      = f2bf(a0[i]);
        fa.u[4 + i]  = f2bf(a1[i]);
        fa.u[8 + i]  = f2bf(a2[i]);
        fa.u[12 + i] = f2bf(a3[i]);
      }
      __builtin_prefetch(pa + 64, 0, 1); // global_prefetch_b8 toward next K tiles
    }

#pragma unroll
    for (int n = 0; n < 4; ++n) {
      Frag fb;
      load_bfrag(fb, &ldsB[(n * 16 + lm) * BSTR + 16 * half]);
      acc[n] = __builtin_amdgcn_wmma_f32_16x16x32_bf16(false, fa.bf, false, fb.bf,
                                                       (short)0, acc[n], false, false);
    }
  }

  const int mh = half * 8; // C layout: lanes 16-31 hold rows M+8
#pragma unroll
  for (int n = 0; n < 4; ++n) {
    const int   col  = tileN + n * 16 + lm;
    const float bcol = bias[col];
    if (OUT_F32) {
      float* C = (float*)Cout;
#pragma unroll
      for (int r = 0; r < 8; ++r)
        C[(size_t)(tileM + r + mh) * N + col] = acc[n][r] + bcol;
    } else {
      unsigned short* C = (unsigned short*)Cout;
#pragma unroll
      for (int r = 0; r < 8; ++r)
        C[(size_t)(tileM + r + mh) * N + col] = f2bf(acc[n][r] + bcol);
    }
  }
}

// ==========================================================================
// Flash attention: one wave (32 threads) per (b, h, 16-query tile).
// Q,K,V in bf16 [B,S,NH*64]; out bf16 [B,S,NH*64]. Causal + pad(id==0) mask.
// ==========================================================================
__global__ __launch_bounds__(32) void attn_kernel(
    const unsigned short* __restrict__ Qb,
    const unsigned short* __restrict__ Kb,
    const unsigned short* __restrict__ Vb,
    const int* __restrict__ ids,
    unsigned short* __restrict__ Ob,
    int S, int NH)
{
  const int QT  = S / 16;
  const int bid = blockIdx.x;
  const int b   = bid / (NH * QT);
  const int rm  = bid % (NH * QT);
  const int h   = rm / QT;
  const int qb16 = (rm % QT) * 16;

  const int lane = threadIdx.x & 31;
  const int lm   = lane & 15;
  const int half = lane >> 4;
  const int kh   = half * 16;
  const int mh   = half * 8;
  const int HD   = NH * 64; // 1024 row stride

  __shared__ __align__(16) unsigned short ldsP[16 * 32];    // probs tile, row-major
  __shared__ __align__(16) unsigned short ldsV[64 * BSTR];  // [dv][key], padded stride

  // Q fragments: dk 0..31 and 32..63 (A layout)
  Frag qa0, qa1;
  {
    const unsigned short* qp = Qb + (size_t)(b * S + qb16 + lm) * HD + h * 64;
    load_afrag(qa0, qp, half);
    load_afrag(qa1, qp + 32, half);
  }

  v8f   o[4] = {v8f{}, v8f{}, v8f{}, v8f{}};
  float mrow[8], lrow[8];
#pragma unroll
  for (int r = 0; r < 8; ++r) { mrow[r] = -__builtin_inff(); lrow[r] = 0.f; }

  const float scale = 0.125f; // 1/sqrt(64)

  for (int kb = 0; kb < qb16 + 16; kb += 32) {
    // ---- scores: two 16-key tiles, DK=64 split in two WMMAs each ----
    v8f sc[2];
#pragma unroll
    for (int t = 0; t < 2; ++t) {
      const unsigned short* kp = Kb + (size_t)(b * S + kb + t * 16 + lm) * HD + h * 64;
      Frag kf0, kf1;
      load_bfrag(kf0, kp + kh);        // dk 0..31 chunk for this lane half
      load_bfrag(kf1, kp + 32 + kh);   // dk 32..63
      v8f s = {};
      s = __builtin_amdgcn_wmma_f32_16x16x32_bf16(false, qa0.bf, false, kf0.bf, (short)0, s, false, false);
      s = __builtin_amdgcn_wmma_f32_16x16x32_bf16(false, qa1.bf, false, kf1.bf, (short)0, s, false, false);
      sc[t] = s;
    }

    // ---- stage V tile (32 keys x 64 dv) transposed into LDS ----
#pragma unroll 8
    for (int i = 0; i < 64; ++i) {
      int e   = lane + 32 * i; // 0..2047
      int key = e >> 6;        // 0..31
      int dv  = e & 63;        // 0..63
      ldsV[dv * BSTR + key] = Vb[(size_t)(b * S + kb + key) * HD + h * 64 + dv];
    }

    // ---- mask (causal + pad) ----
#pragma unroll
    for (int t = 0; t < 2; ++t) {
      const int key = kb + t * 16 + lm;
      const bool pad = (ids[b * S + key] == 0);
#pragma unroll
      for (int r = 0; r < 8; ++r) {
        const int row = qb16 + r + mh;
        float s = sc[t][r] * scale;
        if (pad || key > row) s = -__builtin_inff();
        sc[t][r] = s;
      }
    }

    // ---- online softmax: row reductions across the 16-lane half groups ----
    float alpha[8];
#pragma unroll
    for (int r = 0; r < 8; ++r) {
      float mx = fmaxf(sc[0][r], sc[1][r]);
#pragma unroll
      for (int off = 1; off < 16; off <<= 1)
        mx = fmaxf(mx, __shfl_xor(mx, off, 32));
      const float mnew = fmaxf(mrow[r], mx);
      alpha[r] = __expf(mrow[r] - mnew); // first iter: exp(-inf)=0
      mrow[r]  = mnew;
    }

#pragma unroll
    for (int r = 0; r < 8; ++r) {
      const float m  = mrow[r];
      const float p0 = __expf(sc[0][r] - m);
      const float p1 = __expf(sc[1][r] - m);
      float ls = p0 + p1;
#pragma unroll
      for (int off = 1; off < 16; off <<= 1)
        ls += __shfl_xor(ls, off, 32);
      lrow[r] = lrow[r] * alpha[r] + ls;
      const int row = r + mh;
      ldsP[row * 32 + lm]      = f2bf(p0);
      ldsP[row * 32 + 16 + lm] = f2bf(p1);
#pragma unroll
      for (int n = 0; n < 4; ++n) o[n][r] *= alpha[r];
    }

    __asm__ volatile("s_wait_dscnt 0" ::: "memory"); // LDS writes visible to this wave's reads

    // ---- P(16x32) x V(32x64): 4 WMMAs ----
    Frag fp;
    load_afrag(fp, &ldsP[lm * 32], half);
#pragma unroll
    for (int n = 0; n < 4; ++n) {
      Frag fv;
      load_bfrag(fv, &ldsV[(n * 16 + lm) * BSTR + kh]);
      o[n] = __builtin_amdgcn_wmma_f32_16x16x32_bf16(false, fp.bf, false, fv.bf,
                                                     (short)0, o[n], false, false);
    }
  }

  // ---- normalize and store bf16 ----
#pragma unroll
  for (int r = 0; r < 8; ++r) {
    const float inv = 1.0f / lrow[r];
    const size_t base = (size_t)(b * S + qb16 + r + mh) * HD + h * 64 + lm;
#pragma unroll
    for (int n = 0; n < 4; ++n)
      Ob[base + 16 * n] = f2bf(o[n][r] * inv);
  }
}

// ==========================================================================
extern "C" void kernel_launch(void* const* d_in, const int* in_sizes, int n_in,
                              void* d_out, int out_size, void* d_ws, size_t ws_size,
                              hipStream_t stream)
{
  (void)in_sizes; (void)n_in; (void)out_size; (void)ws_size;
  const int B = 2, S = 2048, H = 1024, NH = 16;
  const int M = B * S; // 4096

  const float* batch = (const float*)d_in[0];
  const int*   ids   = (const int*)  d_in[1];
  const float* W_Q   = (const float*)d_in[2];
  const float* W_K   = (const float*)d_in[3];
  const float* W_V   = (const float*)d_in[4];
  const float* b_Q   = (const float*)d_in[5];
  const float* b_K   = (const float*)d_in[6];
  const float* b_V   = (const float*)d_in[7];
  const float* W_O   = (const float*)d_in[8];
  const float* b_O   = (const float*)d_in[9];

  // workspace: 4 bf16 buffers of [M, H] = 8 MB each (32 MB total)
  unsigned short* Qb = (unsigned short*)d_ws;
  unsigned short* Kb = Qb + (size_t)M * H;
  unsigned short* Vb = Kb + (size_t)M * H;
  unsigned short* Ob = Vb + (size_t)M * H;

  dim3 gg(H / 64, M / 128); // 16 x 32 blocks, 256 threads
  gemm_wmma<false, false><<<gg, 256, 0, stream>>>(batch, W_Q, b_Q, Qb, M, H, H);
  gemm_wmma<false, false><<<gg, 256, 0, stream>>>(batch, W_K, b_K, Kb, M, H, H);
  gemm_wmma<false, false><<<gg, 256, 0, stream>>>(batch, W_V, b_V, Vb, M, H, H);

  attn_kernel<<<B * NH * (S / 16), 32, 0, stream>>>(Qb, Kb, Vb, ids, Ob, S, NH);

  gemm_wmma<true, true><<<gg, 256, 0, stream>>>(Ob, W_O, b_O, (float*)d_out, M, H, H);
}
